// LocallyOptimizedHashingMemory_53163105190610
// MI455X (gfx1250) — matrix operand order repaired and checked by reading, not compiled, not run on hardware
//
#include <hip/hip_runtime.h>
#include <math.h>

// Problem constants (fixed by the reference)
#define BS    8192      // B*T tokens
#define DDIM  1024      // D
#define NH    4         // heads
#define KD    512       // KDIM
#define HALFD 256       // KDIM/2
#define NKEYS 256
#define KNN   32
#define VD    512       // VDIM
#define PKSZ  65536     // NKEYS*NKEYS
#define TQ    512       // BS/16 token tiles

typedef __attribute__((ext_vector_type(8)))  float  v8f;
typedef __attribute__((ext_vector_type(16))) __bf16 v16bf;

// ---------- bf16 helpers: native casts -> v_cvt_pk_bf16_f32-class ops ----------
static __device__ __forceinline__ float  bf2f(__bf16 b) { return (float)b; }
static __device__ __forceinline__ __bf16 f2bf(float f)  { return (__bf16)f; }

// =====================================================================
// Kernel 1: head assignment + bucket tokens by head (one wave / token)
// hd = ||x||^2 - 2 x.c + ||c||^2 ; ||x||^2 constant over h -> skip
// =====================================================================
__global__ __launch_bounds__(256) void k_head(const float* __restrict__ x,
                                              const float* __restrict__ hc,
                                              int* __restrict__ head,
                                              int* __restrict__ bucket,
                                              int* __restrict__ cnt) {
  const int gw   = (blockIdx.x * blockDim.x + threadIdx.x) >> 5;
  const int lane = threadIdx.x & 31;
  if (gw >= BS) return;
  const float* xp = x + (size_t)gw * DDIM;
  float dot[NH] = {0.f, 0.f, 0.f, 0.f};
  float css[NH] = {0.f, 0.f, 0.f, 0.f};
  for (int j = 0; j < DDIM / 32; ++j) {
    float xv = xp[lane + 32 * j];
#pragma unroll
    for (int h = 0; h < NH; ++h) {
      float cv = hc[h * DDIM + lane + 32 * j];
      dot[h] += xv * cv;
      css[h] += cv * cv;
    }
  }
#pragma unroll
  for (int h = 0; h < NH; ++h)
    for (int m = 16; m; m >>= 1) {
      dot[h] += __shfl_xor(dot[h], m);
      css[h] += __shfl_xor(css[h], m);
    }
  float best = 3.0e38f; int bh = 0;
#pragma unroll
  for (int h = 0; h < NH; ++h) {
    float s = css[h] - 2.0f * dot[h];
    if (s < best) { best = s; bh = h; }
  }
  if (lane == 0) {
    head[gw] = bh;
    int pos = atomicAdd(&cnt[bh], 1);
    bucket[bh * BS + pos] = gw;
  }
}

// =====================================================================
// Kernel 2: query GEMM, bucketed by head.
// query[tok, n] = sum_d (x[tok,d]-c_h[d]) * qw[h,n,d] + qb[h,n]
// 256 thr = 8 waves; each wave owns 4 of the 32 column tiles (KD=512).
// bf16 hi/lo split -> 3 x v_wmma_f32_16x16x32_bf16 per (tile, k-chunk).
// =====================================================================
__global__ __launch_bounds__(256) void k_query(const float* __restrict__ x,
                                               const float* __restrict__ hc,
                                               const float* __restrict__ qw,
                                               const float* __restrict__ qb,
                                               const int* __restrict__ bucket,
                                               const int* __restrict__ cnt,
                                               float* __restrict__ query) {
  const int h    = blockIdx.y;
  const int row0 = blockIdx.x * 16;
  const int c    = cnt[h];
  if (row0 >= c) return;

  __shared__ float As[16 * 33];   // padded stride to avoid bank conflicts
  __shared__ int   toks[16];

  const int tid  = threadIdx.x;
  const int lane = tid & 31;
  const int wave = tid >> 5;
  if (tid < 16) {
    int r = row0 + tid;
    toks[tid] = bucket[h * BS + (r < c ? r : row0)];
  }
  const int n   = lane & 15;             // output column within tile / B lane N
  const int akb = (lane < 16) ? 0 : 16;  // A-frag K base
  const int bkb = (lane < 16) ? 0 : 16;  // B-frag K base
  const int m   = n;                     // A-frag row for this lane

  v8f acc[4];
#pragma unroll
  for (int i = 0; i < 4; ++i)
#pragma unroll
    for (int e = 0; e < 8; ++e) acc[i][e] = 0.0f;

  __syncthreads();

  for (int k0 = 0; k0 < DDIM; k0 += 32) {
    // stage (x - c_h) chunk: 16 rows x 32 dims
    for (int i = tid; i < 512; i += 256) {
      int r = i >> 5, cc = i & 31;
      As[r * 33 + cc] = x[(size_t)toks[r] * DDIM + k0 + cc] -
                        hc[(size_t)h * DDIM + k0 + cc];
    }
    __syncthreads();

    // build A fragments (hi/lo)
    v16bf ahi, alo;
#pragma unroll
    for (int e = 0; e < 16; ++e) {
      int K = akb + e;
      float f = As[m * 33 + K];
      __bf16 hb = f2bf(f);
      ahi[e] = hb;
      alo[e] = f2bf(f - bf2f(hb));
    }

#pragma unroll
    for (int i = 0; i < 4; ++i) {
      int n0 = (wave * 4 + i) * 16;
      const float4* bp = reinterpret_cast<const float4*>(
          qw + ((size_t)(h * KD + n0 + n)) * DDIM + k0 + bkb);
      float4 q0 = bp[0], q1 = bp[1], q2 = bp[2], q3 = bp[3];
      float bl[16] = {q0.x, q0.y, q0.z, q0.w, q1.x, q1.y, q1.z, q1.w,
                      q2.x, q2.y, q2.z, q2.w, q3.x, q3.y, q3.z, q3.w};
      v16bf bhi, blo;
#pragma unroll
      for (int e = 0; e < 16; ++e) {
        __bf16 hb = f2bf(bl[e]);
        bhi[e] = hb;
        blo[e] = f2bf(bl[e] - bf2f(hb));
      }
      acc[i] = __builtin_amdgcn_wmma_f32_16x16x32_bf16(false, alo, false, bhi,
                                                       (short)0, acc[i], false, false);
      acc[i] = __builtin_amdgcn_wmma_f32_16x16x32_bf16(false, ahi, false, blo,
                                                       (short)0, acc[i], false, false);
      acc[i] = __builtin_amdgcn_wmma_f32_16x16x32_bf16(false, ahi, false, bhi,
                                                       (short)0, acc[i], false, false);
    }
    __syncthreads();
  }

  // epilogue: + bias, scatter by token id (C/D layout: reg r -> M=r+8*(lane>=16))
  const int mbase = (lane < 16) ? 0 : 8;
#pragma unroll
  for (int i = 0; i < 4; ++i) {
    int n0 = (wave * 4 + i) * 16;
    float bias = qb[h * KD + n0 + n];
#pragma unroll
    for (int r = 0; r < 8; ++r) {
      int tok = toks[r + mbase];
      query[(size_t)tok * KD + n0 + n] = acc[i][r] + bias;
    }
  }
}

// =====================================================================
// Kernel 3: ||q_half||^2 per token and ||key||^2 per (h,half,key)
// =====================================================================
__global__ __launch_bounds__(256) void k_norms(const float* __restrict__ query,
                                               const float* __restrict__ keys,
                                               float* __restrict__ qq,
                                               float* __restrict__ kk) {
  const int wid  = (blockIdx.x * blockDim.x + threadIdx.x) >> 5;
  const int lane = threadIdx.x & 31;
  if (wid < BS) {
    const float* qp = query + (size_t)wid * KD;
    float s0 = 0.f, s1 = 0.f;
    for (int j = 0; j < 8; ++j) {
      float a = qp[lane + 32 * j];          s0 += a * a;
      float b = qp[HALFD + lane + 32 * j];  s1 += b * b;
    }
    for (int m = 16; m; m >>= 1) { s0 += __shfl_xor(s0, m); s1 += __shfl_xor(s1, m); }
    if (lane == 0) { qq[wid * 2] = s0; qq[wid * 2 + 1] = s1; }
  } else {
    int r = wid - BS;
    if (r < NH * 2 * NKEYS) {
      const float* kp = keys + (size_t)r * HALFD;
      float s = 0.f;
      for (int j = 0; j < 8; ++j) { float a = kp[lane + 32 * j]; s += a * a; }
      for (int m = 16; m; m >>= 1) s += __shfl_xor(s, m);
      if (lane == 0) kk[r] = s;
    }
  }
}

// =====================================================================
// Kernel 4: negated sub-distances, bucketed by head. grid=(TQ,NH,2)
// sdist[(tok*2+half), n] = -( qq - 2*q.k + kk ) / KDIM
// 8 waves x 2 column tiles (NKEYS=256). Same 3-product bf16 WMMA.
// =====================================================================
__global__ __launch_bounds__(256) void k_dist(const float* __restrict__ query,
                                              const float* __restrict__ keys,
                                              const int* __restrict__ bucket,
                                              const int* __restrict__ cnt,
                                              const float* __restrict__ qq,
                                              const float* __restrict__ kk,
                                              float* __restrict__ sdist) {
  const int h    = blockIdx.y;
  const int half = blockIdx.z;
  const int row0 = blockIdx.x * 16;
  const int c    = cnt[h];
  if (row0 >= c) return;

  __shared__ float As[16 * 33];
  __shared__ int   toks[16];

  const int tid  = threadIdx.x;
  const int lane = tid & 31;
  const int wave = tid >> 5;
  if (tid < 16) {
    int r = row0 + tid;
    toks[tid] = bucket[h * BS + (r < c ? r : row0)];
  }
  const int n   = lane & 15;
  const int akb = (lane < 16) ? 0 : 16;
  const int bkb = (lane < 16) ? 0 : 16;
  const int m   = n;

  v8f acc[2];
#pragma unroll
  for (int i = 0; i < 2; ++i)
#pragma unroll
    for (int e = 0; e < 8; ++e) acc[i][e] = 0.0f;

  __syncthreads();

  for (int k0 = 0; k0 < HALFD; k0 += 32) {
    for (int i = tid; i < 512; i += 256) {
      int r = i >> 5, cc = i & 31;
      As[r * 33 + cc] = query[(size_t)toks[r] * KD + half * HALFD + k0 + cc];
    }
    __syncthreads();

    v16bf ahi, alo;
#pragma unroll
    for (int e = 0; e < 16; ++e) {
      int K = akb + e;
      float f = As[m * 33 + K];
      __bf16 hb = f2bf(f);
      ahi[e] = hb;
      alo[e] = f2bf(f - bf2f(hb));
    }

#pragma unroll
    for (int i = 0; i < 2; ++i) {
      int n0 = (wave * 2 + i) * 16;
      const float4* bp = reinterpret_cast<const float4*>(
          keys + ((size_t)((h * 2 + half) * NKEYS + n0 + n)) * HALFD + k0 + bkb);
      float4 q0 = bp[0], q1 = bp[1], q2 = bp[2], q3 = bp[3];
      float bl[16] = {q0.x, q0.y, q0.z, q0.w, q1.x, q1.y, q1.z, q1.w,
                      q2.x, q2.y, q2.z, q2.w, q3.x, q3.y, q3.z, q3.w};
      v16bf bhi, blo;
#pragma unroll
      for (int e = 0; e < 16; ++e) {
        __bf16 hb = f2bf(bl[e]);
        bhi[e] = hb;
        blo[e] = f2bf(bl[e] - bf2f(hb));
      }
      acc[i] = __builtin_amdgcn_wmma_f32_16x16x32_bf16(false, alo, false, bhi,
                                                       (short)0, acc[i], false, false);
      acc[i] = __builtin_amdgcn_wmma_f32_16x16x32_bf16(false, ahi, false, blo,
                                                       (short)0, acc[i], false, false);
      acc[i] = __builtin_amdgcn_wmma_f32_16x16x32_bf16(false, ahi, false, bhi,
                                                       (short)0, acc[i], false, false);
    }
    __syncthreads();
  }

  const int mbase = (lane < 16) ? 0 : 8;
  const float invk = 1.0f / (float)KD;
#pragma unroll
  for (int i = 0; i < 2; ++i) {
    int n0 = (wave * 2 + i) * 16;
    float kkn = kk[(h * 2 + half) * NKEYS + n0 + n];
#pragma unroll
    for (int r = 0; r < 8; ++r) {
      int tok = toks[r + mbase];
      float d = (qq[tok * 2 + half] - 2.0f * acc[i][r] + kkn) * invk;
      sdist[((size_t)tok * 2 + half) * NKEYS + n0 + n] = -d;
    }
  }
}

// =====================================================================
// Kernel 5: top-32 of 256 per (token, half). One wave per (token,half).
// Iterative select-max with (value desc, index asc) ordering.
// =====================================================================
__global__ __launch_bounds__(256) void k_topk(const float* __restrict__ sdist,
                                              float* __restrict__ stopv,
                                              int* __restrict__ itop) {
  const int wid  = (blockIdx.x * blockDim.x + threadIdx.x) >> 5;
  const int lane = threadIdx.x & 31;
  if (wid >= BS * 2) return;
  const float* base = sdist + (size_t)wid * NKEYS;
  float v[8];
#pragma unroll
  for (int i = 0; i < 8; ++i) v[i] = base[lane * 8 + i];

  for (int sel = 0; sel < KNN; ++sel) {
    float bv = -3.0e38f; int bi = 0x7FFFFFFF;
#pragma unroll
    for (int i = 0; i < 8; ++i) {
      if (v[i] > bv) { bv = v[i]; bi = lane * 8 + i; }
    }
    for (int mm = 16; mm; mm >>= 1) {
      float ov = __shfl_xor(bv, mm);
      int   oi = __shfl_xor(bi, mm);
      if (ov > bv || (ov == bv && oi < bi)) { bv = ov; bi = oi; }
    }
    if (lane == 0) {
      stopv[(size_t)wid * KNN + sel] = bv;
      itop[(size_t)wid * KNN + sel] = bi;
    }
    if ((bi >> 3) == lane) {
      int ii = bi & 7;
#pragma unroll
      for (int i = 0; i < 8; ++i) if (i == ii) v[i] = -3.0e38f;
    }
  }
}

// =====================================================================
// Kernel 6: combine halves, top-32 of 32x32, softmax, values gather.
// One 256-thread block per token; wave0 does selection; all waves gather.
// =====================================================================
__global__ __launch_bounds__(256) void k_out(const float* __restrict__ stopv,
                                             const int* __restrict__ itop,
                                             const int* __restrict__ head,
                                             const float* __restrict__ vals,
                                             const float* __restrict__ dw,
                                             float* __restrict__ out) {
  const int t    = blockIdx.x;
  const int tid  = threadIdx.x;
  const int lane = tid & 31;
  const int wave = tid >> 5;
  const int h    = head[t];

  __shared__ float ps[KNN];
  __shared__ int   gs[KNN];

  if (wave == 0) {
    const size_t b1 = (size_t)(2 * t) * KNN;
    const size_t b2 = (size_t)(2 * t + 1) * KNN;
    float s1v = stopv[b1 + lane];  int i1v = itop[b1 + lane];
    float s2v = stopv[b2 + lane];  int i2v = itop[b2 + lane];
    unsigned used = 0u;         // bit i: combo (i, j=lane) consumed
    float myb = 0.f; int mygi = 0;

    for (int sel = 0; sel < KNN; ++sel) {
      float bv = -3.0e38f; int bidx = 0x7FFFFFFF;
#pragma unroll
      for (int i = 0; i < KNN; ++i) {
        float vv = __shfl(s1v, i) + s2v;
        bool ok = ((used >> i) & 1u) == 0u;
        int idx = i * KNN + lane;
        if (ok && (vv > bv || (vv == bv && idx < bidx))) { bv = vv; bidx = idx; }
      }
      for (int mm = 16; mm; mm >>= 1) {
        float ov = __shfl_xor(bv, mm);
        int   oi = __shfl_xor(bidx, mm);
        if (ov > bv || (ov == bv && oi < bidx)) { bv = ov; bidx = oi; }
      }
      int bi = bidx >> 5;   // i1 slot  (idx = i*32 + j)
      int bj = bidx & 31;   // i2 slot
      int gi = __shfl(i1v, bi) * NKEYS + __shfl(i2v, bj);
      if (lane == sel) { myb = bv; mygi = gi; }
      if (lane == bj) used |= (1u << bi);
    }

    float dwv = dw[h];
    float sc  = -sqrtf(fmaxf(-myb, 0.f)) * dwv;
    float mx  = __shfl(sc, 0);          // sel 0 is the max (sorted desc)
    float ex  = __expf(sc - mx);
    float sm  = ex;
    for (int mm = 16; mm; mm >>= 1) sm += __shfl_xor(sm, mm);
    ps[lane] = ex / sm;
    gs[lane] = mygi + h * PKSZ;
  }
  __syncthreads();

  float a0 = 0.f, a1 = 0.f;
  for (int r = 0; r < KNN; ++r) {
    float p = ps[r];
    size_t base = (size_t)gs[r] * VD;
    a0 += p * vals[base + tid];
    a1 += p * vals[base + tid + 256];
  }
  out[(size_t)t * VD + tid]       = a0;
  out[(size_t)t * VD + tid + 256] = a1;
}

// =====================================================================
extern "C" void kernel_launch(void* const* d_in, const int* in_sizes, int n_in,
                              void* d_out, int out_size, void* d_ws, size_t ws_size,
                              hipStream_t stream) {
  const float* x    = (const float*)d_in[0];
  const float* hc   = (const float*)d_in[1];
  const float* qw   = (const float*)d_in[2];
  const float* qb   = (const float*)d_in[3];
  const float* keys = (const float*)d_in[4];
  const float* vals = (const float*)d_in[5];
  const float* dw   = (const float*)d_in[6];
  float* out = (float*)d_out;

  char* w = (char*)d_ws;
  auto take = [&](size_t bytes) -> char* {
    char* p = w;
    w += (bytes + 255) & ~(size_t)255;
    return p;
  };
  int*   cnt    = (int*)  take(NH * sizeof(int));
  int*   head   = (int*)  take((size_t)BS * sizeof(int));
  int*   bucket = (int*)  take((size_t)NH * BS * sizeof(int));
  float* query  = (float*)take((size_t)BS * KD * sizeof(float));
  float* qq     = (float*)take((size_t)BS * 2 * sizeof(float));
  float* kk     = (float*)take((size_t)NH * 2 * NKEYS * sizeof(float));
  float* sdist  = (float*)take((size_t)BS * 2 * NKEYS * sizeof(float));
  float* stopv  = (float*)take((size_t)BS * 2 * KNN * sizeof(float));
  int*   itop   = (int*)  take((size_t)BS * 2 * KNN * sizeof(int));

  hipMemsetAsync(cnt, 0, NH * sizeof(int), stream);
  k_head <<<BS / 8, 256, 0, stream>>>(x, hc, head, bucket, cnt);
  k_query<<<dim3(TQ, NH), 256, 0, stream>>>(x, hc, qw, qb, bucket, cnt, query);
  k_norms<<<(BS / 8) + (NH * 2 * NKEYS) / 8, 256, 0, stream>>>(query, keys, qq, kk);
  k_dist <<<dim3(TQ, NH, 2), 256, 0, stream>>>(query, keys, bucket, cnt, qq, kk, sdist);
  k_topk <<<(BS * 2) / 8, 256, 0, stream>>>(sdist, stopv, itop);
  k_out  <<<BS, 256, 0, stream>>>(stopv, itop, head, vals, dw, out);

  (void)in_sizes; (void)n_in; (void)out_size; (void)ws_size;
}
